// Prompt_Encoder_43413529428592
// MI455X (gfx1250) — matrix operand order repaired and checked by reading, compiled, not run on hardware
//
#include <hip/hip_runtime.h>
#include <math.h>

// ---------------- problem constants ----------------
constexpr int D    = 300;          // feature dim
constexpr int H    = 5;            // heads
constexpr int N    = 4096;         // nodes / graph
constexpr int E    = 65536;        // edges / graph
constexpr int B    = 4;            // graphs
constexpr int NOUT = H * D;        // 1500 (GEMM N dim)

constexpr int KPAD = 320;          // K padded to mult of 32
constexpr int KT   = KPAD / 32;    // 10 k-steps
constexpr int NPAD = 1536;         // N padded to mult of 64
constexpr int NT   = NPAD / 16;    // 96 n-tiles
constexpr int MT   = N / 16;       // 256 m-tiles
constexpr int NTW  = 4;            // n-tiles per wave (A-fragment reuse)
constexpr int NG   = NT / NTW;     // 24 n-groups

typedef __bf16 v16bf __attribute__((ext_vector_type(16)));
typedef float  v8f   __attribute__((ext_vector_type(8)));

// ---------------- operand pre-swizzle ----------------
// B-fragment layout (32x16 K-major, bf16): lane = half*16+col, element j -> K = half*16 + j
__global__ void pack_w_kernel(const float* __restrict__ W, __bf16* __restrict__ Wp) {
  int idx = blockIdx.x * blockDim.x + threadIdx.x;     // over KT*NT*32 lane-slots
  if (idx >= KT * NT * 32) return;
  int lane = idx & 31;
  int nt   = (idx >> 5) % NT;
  int kt   = (idx >> 5) / NT;
  int half = lane >> 4, col = lane & 15;
  int n = nt * 16 + col;
  __bf16* dst = Wp + (size_t)idx * 16;
#pragma unroll
  for (int j = 0; j < 16; ++j) {
    int k = kt * 32 + half * 16 + j;
    float v = (k < D && n < NOUT) ? W[(size_t)k * NOUT + n] : 0.0f;
    dst[j] = (__bf16)v;
  }
}

// A-fragment layout (16x32 M-major, bf16): lane = half*16+row,
// element j -> K = half*16 + j
__global__ void pack_a_kernel(const float* __restrict__ x, __bf16* __restrict__ Apk) {
  int idx = blockIdx.x * blockDim.x + threadIdx.x;     // over MT*KT*32 lane-slots
  if (idx >= MT * KT * 32) return;
  int lane = idx & 31;
  int kt   = (idx >> 5) % KT;
  int mt   = (idx >> 5) / KT;
  int half = lane >> 4, row = lane & 15;
  int m = mt * 16 + row;
  __bf16* dst = Apk + (size_t)idx * 16;
#pragma unroll
  for (int j = 0; j < 16; ++j) {
    int p = j >> 1, lo = j & 1;
    int kl = half * 16 + j;
    int k = kt * 32 + kl;
    float v = (k < D) ? x[(size_t)m * D + k] : 0.0f;
    dst[j] = (__bf16)v;
  }
}

// ---------------- WMMA GEMM: h[N,NPAD] = A[N,KPAD] * W[KPAD,NPAD] ----------------
// Register double-buffered: loads for k-step kt+1 are issued before the 4 WMMAs
// of step kt, so load latency is hidden behind matrix issue instead of
// s_wait_loadcnt stalls directly in front of each WMMA.
__global__ void __launch_bounds__(256) gemm_kernel(const __bf16* __restrict__ Apk,
                                                   const __bf16* __restrict__ Wpk,
                                                   float* __restrict__ Hh) {
  const v16bf* Af = (const v16bf*)Apk;
  const v16bf* Bf = (const v16bf*)Wpk;
  int lane = threadIdx.x & 31;
  int gw   = blockIdx.x * 8 + (threadIdx.x >> 5);      // global wave id
  int mt = gw / NG, ng = gw % NG;
  int abase = mt * KT;
  int nb    = ng * NTW;

  // current-step fragments
  v16bf a  = Af[(size_t)abase * 32 + lane];
  v16bf b0 = Bf[(size_t)(nb + 0) * 32 + lane];
  v16bf b1 = Bf[(size_t)(nb + 1) * 32 + lane];
  v16bf b2 = Bf[(size_t)(nb + 2) * 32 + lane];
  v16bf b3 = Bf[(size_t)(nb + 3) * 32 + lane];

  v8f c0 = {}, c1 = {}, c2 = {}, c3 = {};
#pragma unroll
  for (int kt = 0; kt < KT - 1; ++kt) {
    // issue next-step loads first (double buffer)
    v16bf an  = Af[(size_t)(abase + kt + 1) * 32 + lane];
    v16bf bn0 = Bf[(size_t)((kt + 1) * NT + nb + 0) * 32 + lane];
    v16bf bn1 = Bf[(size_t)((kt + 1) * NT + nb + 1) * 32 + lane];
    v16bf bn2 = Bf[(size_t)((kt + 1) * NT + nb + 2) * 32 + lane];
    v16bf bn3 = Bf[(size_t)((kt + 1) * NT + nb + 3) * 32 + lane];
    // consume current-step fragments
    c0 = __builtin_amdgcn_wmma_f32_16x16x32_bf16(false, a, false, b0, (short)0, c0, false, false);
    c1 = __builtin_amdgcn_wmma_f32_16x16x32_bf16(false, a, false, b1, (short)0, c1, false, false);
    c2 = __builtin_amdgcn_wmma_f32_16x16x32_bf16(false, a, false, b2, (short)0, c2, false, false);
    c3 = __builtin_amdgcn_wmma_f32_16x16x32_bf16(false, a, false, b3, (short)0, c3, false, false);
    a = an; b0 = bn0; b1 = bn1; b2 = bn2; b3 = bn3;
  }
  // tail step
  c0 = __builtin_amdgcn_wmma_f32_16x16x32_bf16(false, a, false, b0, (short)0, c0, false, false);
  c1 = __builtin_amdgcn_wmma_f32_16x16x32_bf16(false, a, false, b1, (short)0, c1, false, false);
  c2 = __builtin_amdgcn_wmma_f32_16x16x32_bf16(false, a, false, b2, (short)0, c2, false, false);
  c3 = __builtin_amdgcn_wmma_f32_16x16x32_bf16(false, a, false, b3, (short)0, c3, false, false);

  // C/D layout: lane = half*16+col, VGPR r -> M = r + 8*half
  int half = lane >> 4, col = lane & 15;
  int mrow = mt * 16 + 8 * half;
  v8f cs[4] = {c0, c1, c2, c3};
#pragma unroll
  for (int t = 0; t < NTW; ++t) {
    int n = (nb + t) * 16 + col;
    if (n < NOUT) {
#pragma unroll
      for (int r = 0; r < 8; ++r)
        Hh[(size_t)(mrow + r) * NPAD + n] = cs[t][r];
    }
  }
}

// ---------------- attention scores: a_src/a_dst [N,H] ----------------
__global__ void __launch_bounds__(256) attn_kernel(const float* __restrict__ Hh,
                                                   const float* __restrict__ att_s,
                                                   const float* __restrict__ att_d,
                                                   float* __restrict__ osrc,
                                                   float* __restrict__ odst) {
  int lane = threadIdx.x & 31;
  int gw   = blockIdx.x * 8 + (threadIdx.x >> 5);
  if (gw >= N * H) return;
  int n = gw / H, hh = gw % H;
  const float* hp = Hh + (size_t)n * NPAD + hh * D;
  const float* sp = att_s + hh * D;
  const float* dp = att_d + hh * D;
  float s0 = 0.f, s1 = 0.f;
  for (int c = lane; c < D; c += 32) { float v = hp[c]; s0 += v * sp[c]; s1 += v * dp[c]; }
  for (int off = 16; off; off >>= 1) { s0 += __shfl_xor(s0, off, 32); s1 += __shfl_xor(s1, off, 32); }
  if (lane == 0) { osrc[gw] = s0; odst[gw] = s1; }
}

__global__ void fill_kernel(float* __restrict__ p, int n, float v) {
  int i = blockIdx.x * blockDim.x + threadIdx.x;
  if (i < n) p[i] = v;
}

__device__ __forceinline__ void atomicMaxFloat(float* addr, float v) {
  if (v >= 0.f) atomicMax((int*)addr, __float_as_int(v));
  else          atomicMin((unsigned int*)addr, __float_as_uint(v));
}

// ---------------- edge phase ----------------
__global__ void edge_max_kernel(const long long* __restrict__ ei,
                                const float* __restrict__ asrc, const float* __restrict__ adst,
                                float* __restrict__ alpha, float* __restrict__ mx) {
  int t = blockIdx.x * blockDim.x + threadIdx.x;
  if (t >= E * H) return;
  int e = t / H, hh = t % H;
  int s = (int)ei[e], d = (int)ei[E + e];
  float a = asrc[s * H + hh] + adst[d * H + hh];
  a = a > 0.f ? a : 0.2f * a;                       // leaky_relu(0.2)
  alpha[t] = a;
  atomicMaxFloat(&mx[d * H + hh], a);
}

__global__ void edge_exp_kernel(const long long* __restrict__ ei,
                                float* __restrict__ alpha, const float* __restrict__ mx,
                                float* __restrict__ den) {
  int t = blockIdx.x * blockDim.x + threadIdx.x;
  if (t >= E * H) return;
  int e = t / H, hh = t % H;
  int d = (int)ei[E + e];
  float a = expf(alpha[t] - mx[d * H + hh]);
  alpha[t] = a;
  atomicAdd(&den[d * H + hh], a);
}

// wave per edge; fuse head-mean into the scatter (atomicAdd into [N,D] only)
__global__ void __launch_bounds__(256) edge_agg_kernel(const long long* __restrict__ ei,
                                                       const float* __restrict__ Hh,
                                                       const float* __restrict__ alpha,
                                                       const float* __restrict__ den,
                                                       float* __restrict__ outm) {
  int lane = threadIdx.x & 31;
  int e = blockIdx.x * 8 + (threadIdx.x >> 5);
  if (e >= E) return;
  int s = (int)ei[e], d = (int)ei[E + e];
  float w[H];
#pragma unroll
  for (int hh = 0; hh < H; ++hh)
    w[hh] = alpha[(size_t)e * H + hh] / (den[d * H + hh] + 1e-16f) * (1.0f / H);
  const float* hp = Hh + (size_t)s * NPAD;
  float* op = outm + (size_t)d * D;
  for (int c = lane; c < D; c += 32) {
    float acc = 0.f;
#pragma unroll
    for (int hh = 0; hh < H; ++hh) acc += w[hh] * hp[hh * D + c];
    atomicAdd(&op[c], acc);
  }
}

// ---------------- bias + exact GELU + LayerNorm, wave per node ----------------
__global__ void __launch_bounds__(256) finalize_kernel(const float* __restrict__ outm,
                                                       const float* __restrict__ bias,
                                                       const float* __restrict__ gamma,
                                                       const float* __restrict__ beta,
                                                       float* __restrict__ out) {
  int lane = threadIdx.x & 31;
  int n = blockIdx.x * 8 + (threadIdx.x >> 5);
  if (n >= N) return;
  const float* ip = outm + (size_t)n * D;
  float g[10];
  float s = 0.f, s2 = 0.f;
  int i = 0;
  for (int c = lane; c < D; c += 32, ++i) {
    float x = ip[c] + bias[c];
    float ge = 0.5f * x * (1.0f + erff(x * 0.70710678118654752440f));
    g[i] = ge; s += ge; s2 += ge * ge;
  }
  for (int off = 16; off; off >>= 1) { s += __shfl_xor(s, off, 32); s2 += __shfl_xor(s2, off, 32); }
  float mu  = s * (1.0f / D);
  float var = s2 * (1.0f / D) - mu * mu;
  float rs  = rsqrtf(var + 1e-5f);
  float* op = out + (size_t)n * D;
  i = 0;
  for (int c = lane; c < D; c += 32, ++i)
    op[c] = (g[i] - mu) * rs * gamma[c] + beta[c];
}

// ---------------- host orchestration ----------------
extern "C" void kernel_launch(void* const* d_in, const int* in_sizes, int n_in,
                              void* d_out, int out_size, void* d_ws, size_t ws_size,
                              hipStream_t stream) {
  (void)in_sizes; (void)n_in; (void)out_size; (void)ws_size;
  const float*     v2     = (const float*)d_in[0];
  const long long* ei_all = (const long long*)d_in[1];
  const float* Wl[2]  = {(const float*)d_in[2], (const float*)d_in[6]};
  const float* asl[2] = {(const float*)d_in[3], (const float*)d_in[7]};
  const float* adl[2] = {(const float*)d_in[4], (const float*)d_in[8]};
  const float* bl[2]  = {(const float*)d_in[5], (const float*)d_in[9]};
  const float* gamma  = (const float*)d_in[10];
  const float* beta   = (const float*)d_in[11];
  float* out = (float*)d_out;

  // carve workspace
  char* ws = (char*)d_ws;
  size_t off = 0;
  auto carve = [&](size_t bytes) -> void* {
    void* p = ws + off;
    off += (bytes + 255) & ~(size_t)255;
    return p;
  };
  __bf16* Wp   = (__bf16*)carve((size_t)KT * NT * 32 * 16 * sizeof(__bf16));   // ~0.96 MB
  __bf16* Apk  = (__bf16*)carve((size_t)MT * KT * 32 * 16 * sizeof(__bf16));   // ~2.6 MB
  float*  Hh   = (float*)carve((size_t)N * NPAD * sizeof(float));              // ~25 MB
  float*  asrc = (float*)carve((size_t)N * H * sizeof(float));
  float*  adst = (float*)carve((size_t)N * H * sizeof(float));
  float*  mx   = (float*)carve((size_t)N * H * sizeof(float));
  float*  den  = (float*)carve((size_t)N * H * sizeof(float));
  float*  alph = (float*)carve((size_t)E * H * sizeof(float));                 // ~1.3 MB
  float*  outm = (float*)carve((size_t)N * D * sizeof(float));                 // ~4.9 MB
  float*  X1   = (float*)carve((size_t)B * N * D * sizeof(float));             // ~19.7 MB

  for (int layer = 0; layer < 2; ++layer) {
    pack_w_kernel<<<(KT * NT * 32 + 255) / 256, 256, 0, stream>>>(Wl[layer], Wp);
    for (int b = 0; b < B; ++b) {
      const float*     xin  = (layer == 0 ? v2 : X1) + (size_t)b * N * D;
      const long long* ei   = ei_all + (size_t)b * 2 * E;
      float*           yout = (layer == 0 ? X1 : out) + (size_t)b * N * D;

      pack_a_kernel<<<(MT * KT * 32 + 255) / 256, 256, 0, stream>>>(xin, Apk);
      gemm_kernel<<<(MT * NG) / 8, 256, 0, stream>>>(Apk, Wp, Hh);
      attn_kernel<<<(N * H) / 8, 256, 0, stream>>>(Hh, asl[layer], adl[layer], asrc, adst);

      fill_kernel<<<(N * H + 255) / 256, 256, 0, stream>>>(mx, N * H, -INFINITY);
      fill_kernel<<<(N * H + 255) / 256, 256, 0, stream>>>(den, N * H, 0.0f);
      fill_kernel<<<(N * D + 255) / 256, 256, 0, stream>>>(outm, N * D, 0.0f);

      edge_max_kernel<<<(E * H + 255) / 256, 256, 0, stream>>>(ei, asrc, adst, alph, mx);
      edge_exp_kernel<<<(E * H + 255) / 256, 256, 0, stream>>>(ei, alph, mx, den);
      edge_agg_kernel<<<E / 8, 256, 0, stream>>>(ei, Hh, alph, den, outm);
      finalize_kernel<<<N / 8, 256, 0, stream>>>(outm, bl[layer], gamma, beta, yout);
    }
  }
}
